// VishwamAIModel_65798898974868
// MI455X (gfx1250) — compile-verified, matrix operand order checked by reading, not run on hardware
//
#include <hip/hip_runtime.h>
#include <hip/hip_bf16.h>
#include <stdint.h>

// ---------- types ----------
typedef __attribute__((ext_vector_type(16))) __bf16 v16bf;
typedef __attribute__((ext_vector_type(8)))  float  v8f;
typedef __attribute__((ext_vector_type(4)))  unsigned int v4u;
typedef unsigned short u16t;
typedef unsigned int   u32t;

#define TOK_N  8192   // B*S
#define SEQ    1024
#define BATCH  8
#define HEADS  8
#define KSZ    32
#define MODEL  256

union FragBF {
  v16bf v;
  v4u   q[2];
};

__device__ __forceinline__ u16t f2bf(float f) {
  u32t u = __builtin_bit_cast(u32t, f);
  u += 0x7FFFu + ((u >> 16) & 1u);   // round-to-nearest-even
  return (u16t)(u >> 16);
}
__device__ __forceinline__ float bf2f(u16t h) {
  u32t u = ((u32t)h) << 16;
  return __builtin_bit_cast(float, u);
}

// A/B fragment for v_wmma_f32_16x16x32_bf16 from row-major [16 x K] data
// lane<16 : row = lane,    K chunks [0..7] and [16..23]
// lane>=16: row = lane-16, K chunks [8..15] and [24..31]
__device__ __forceinline__ void load_frag(FragBF& f, const u16t* base, int stride, int lane) {
  int r  = lane & 15;
  int c0 = (lane < 16) ? 0 : 8;
  const u16t* p = base + (size_t)r * stride + c0;
  f.q[0] = *(const v4u*)(p);
  f.q[1] = *(const v4u*)(p + 16);
}

__device__ __forceinline__ v8f wmma_bf16(const FragBF& a, const FragBF& b, v8f c) {
  return __builtin_amdgcn_wmma_f32_16x16x32_bf16(false, a.v, false, b.v, (short)0, c, false, false);
}

// ---------- weight transpose + bf16 convert: W[K,N] f32 -> Wt[N,K] bf16 ----------
__global__ __launch_bounds__(256) void vw_transw(const float* __restrict__ W,
                                                 u16t* __restrict__ Wt, int K, int N) {
  int i = blockIdx.x * 256 + threadIdx.x;
  if (i >= K * N) return;
  int n = i / K, kk = i - n * K;
  Wt[i] = f2bf(W[(size_t)kk * N + n]);
}

// ---------- embedding gather -> bf16 ----------
__global__ __launch_bounds__(256) void vw_embed(const int* __restrict__ tok,
                                                const float* __restrict__ We,
                                                u16t* __restrict__ emb) {
  int i = blockIdx.x * 256 + threadIdx.x;   // < TOK_N*128
  int t = i >> 7, c = i & 127;
  emb[i] = f2bf(We[(size_t)tok[t] * 128 + c]);
}

// ---------- register-blocked GEMM: C[M,N] = A[M,K](bf16) @ Wt[N,K](bf16)^T + bias ----
// Each wave computes a 32(M) x 64(N) strip: 2 A-frags x 4 B-frags -> 8 WMMAs per k-step
// mode 0: store bf16 row-major [M,N]
// mode 1: out(float*)[M,N] += result   (expert accumulation)
// mode 2: store bf16 transposed V layout [b, h, d, s]  (M must be B*S, N = MODEL)
__global__ __launch_bounds__(256) void vw_gemm(const u16t* __restrict__ A,
                                               const u16t* __restrict__ Wt,
                                               const float* __restrict__ bias,
                                               void* __restrict__ out,
                                               int M, int N, int K, int mode) {
  int lane = threadIdx.x & 31;
  int wave = threadIdx.x >> 5;
  int tile = blockIdx.x * 8 + wave;
  int nstrips = N >> 6;                 // strips of 64 columns
  int ntiles  = (M >> 5) * nstrips;     // strips of 32 rows
  if (tile >= ntiles) return;
  int mt = tile / nstrips;
  int nt = tile - mt * nstrips;

  const u16t* Ab = A  + (size_t)(mt * 32) * K;
  const u16t* Bb = Wt + (size_t)(nt * 64) * K;

  v8f acc[2][4];
#pragma unroll
  for (int mi = 0; mi < 2; ++mi)
#pragma unroll
    for (int ni = 0; ni < 4; ++ni) acc[mi][ni] = (v8f){};

  for (int k0 = 0; k0 < K; k0 += 32) {
    FragBF af[2], bf[4];
#pragma unroll
    for (int mi = 0; mi < 2; ++mi)
      load_frag(af[mi], Ab + (size_t)(mi * 16) * K + k0, K, lane);
#pragma unroll
    for (int ni = 0; ni < 4; ++ni)
      load_frag(bf[ni], Bb + (size_t)(ni * 16) * K + k0, K, lane);

    if (k0 + 32 < K) {   // pull next k-slab toward the WGP while WMMAs run
      __builtin_prefetch(Ab + (size_t)(lane & 15) * K + k0 + 32, 0, 3);
      __builtin_prefetch(Bb + (size_t)(lane & 15) * K + k0 + 32, 0, 3);
    }

#pragma unroll
    for (int mi = 0; mi < 2; ++mi)
#pragma unroll
      for (int ni = 0; ni < 4; ++ni)
        acc[mi][ni] = wmma_bf16(af[mi], bf[ni], acc[mi][ni]);
  }

  int lane15 = lane & 15;
  int g = (lane >> 4) * 8;   // row group offset (C layout: lanes 16-31 hold rows 8-15)

  float bvv[4];
#pragma unroll
  for (int ni = 0; ni < 4; ++ni)
    bvv[ni] = bias ? bias[nt * 64 + ni * 16 + lane15] : 0.0f;

  if (mode == 0) {
    u16t* O = (u16t*)out;
#pragma unroll
    for (int mi = 0; mi < 2; ++mi)
#pragma unroll
      for (int ni = 0; ni < 4; ++ni) {
        int n = nt * 64 + ni * 16 + lane15;
#pragma unroll
        for (int j = 0; j < 8; ++j) {
          int m = mt * 32 + mi * 16 + j + g;
          O[(size_t)m * N + n] = f2bf(acc[mi][ni][j] + bvv[ni]);
        }
      }
  } else if (mode == 1) {
    float* O = (float*)out;
#pragma unroll
    for (int mi = 0; mi < 2; ++mi)
#pragma unroll
      for (int ni = 0; ni < 4; ++ni) {
        int n = nt * 64 + ni * 16 + lane15;
#pragma unroll
        for (int j = 0; j < 8; ++j) {
          int m = mt * 32 + mi * 16 + j + g;
          O[(size_t)m * N + n] += acc[mi][ni][j] + bvv[ni];
        }
      }
  } else {
    u16t* O = (u16t*)out;
#pragma unroll
    for (int mi = 0; mi < 2; ++mi)
#pragma unroll
      for (int ni = 0; ni < 4; ++ni) {
        int n = nt * 64 + ni * 16 + lane15;
        int h = n >> 5, d = n & 31;
#pragma unroll
        for (int j = 0; j < 8; ++j) {
          int m = mt * 32 + mi * 16 + j + g;
          int b = m >> 10, s = m & (SEQ - 1);
          O[(size_t)((b * HEADS + h) * KSZ + d) * SEQ + s] = f2bf(acc[mi][ni][j] + bvv[ni]);
        }
      }
  }
}

// ---------- flash attention: q,k [T,256] bf16, vT [b,h,d,s] bf16 -> o [T,256] bf16 ----------
// one wave per 16-query tile; 4 waves / block
__global__ __launch_bounds__(128) void vw_attn(const u16t* __restrict__ q,
                                               const u16t* __restrict__ k,
                                               const u16t* __restrict__ vT,
                                               u16t* __restrict__ o) {
  __shared__ u16t plds[4][16 * 32];

  int lane = threadIdx.x & 31;
  int wave = threadIdx.x >> 5;
  int qt = blockIdx.x * 4 + wave;          // 0 .. 4095
  int b   = qt >> 9;                        // 512 qtiles per batch (8 heads * 64)
  int rem = qt & 511;
  int h   = rem >> 6;
  int q0  = (rem & 63) << 4;
  const float scale = 0.17677669529663687f; // 1/sqrt(32)

  FragBF qf;
  {
    int r = lane & 15, c0 = (lane < 16) ? 0 : 8;
    const u16t* p = q + ((size_t)(b * SEQ + q0 + r)) * MODEL + h * KSZ + c0;
    qf.q[0] = *(const v4u*)p;
    qf.q[1] = *(const v4u*)(p + 16);
  }

  v8f o0 = {}, o1 = {};
  float rmax[8], rsum[8];
#pragma unroll
  for (int j = 0; j < 8; ++j) { rmax[j] = -3.0e38f; rsum[j] = 0.f; }

  const u16t* vbase = vT + ((size_t)(b * HEADS + h)) * KSZ * SEQ;
  u16t* L = plds[wave];

  for (int k0 = 0; k0 < SEQ; k0 += 32) {
    // K fragments: B operand of q*k^T -> lane n holds key (k0 + sub*16 + n), contiguous d
    FragBF kf0, kf1;
    {
      int n = lane & 15, c0 = (lane < 16) ? 0 : 8;
      const u16t* p0 = k + ((size_t)(b * SEQ + k0 + n)) * MODEL + h * KSZ + c0;
      kf0.q[0] = *(const v4u*)p0;
      kf0.q[1] = *(const v4u*)(p0 + 16);
      const u16t* p1 = p0 + (size_t)16 * MODEL;
      kf1.q[0] = *(const v4u*)p1;
      kf1.q[1] = *(const v4u*)(p1 + 16);
    }
    v8f z = {};
    v8f s0 = wmma_bf16(qf, kf0, z);
    v8f s1 = wmma_bf16(qf, kf1, z);

    float bm[8];
#pragma unroll
    for (int j = 0; j < 8; ++j) {
      s0[j] *= scale; s1[j] *= scale;
      bm[j] = fmaxf(s0[j], s1[j]);
    }
    // row max across the 16-lane group holding this row
#pragma unroll
    for (int mset = 1; mset < 16; mset <<= 1)
#pragma unroll
      for (int j = 0; j < 8; ++j)
        bm[j] = fmaxf(bm[j], __shfl_xor(bm[j], mset, 32));

    float p0a[8], p1a[8], bs[8];
#pragma unroll
    for (int j = 0; j < 8; ++j) {
      float nm   = fmaxf(rmax[j], bm[j]);
      float corr = __expf(rmax[j] - nm);
      p0a[j] = __expf(s0[j] - nm);
      p1a[j] = __expf(s1[j] - nm);
      bs[j]  = p0a[j] + p1a[j];
      o0[j] *= corr; o1[j] *= corr;
      rsum[j] *= corr;
      rmax[j] = nm;
    }
#pragma unroll
    for (int mset = 1; mset < 16; mset <<= 1)
#pragma unroll
      for (int j = 0; j < 8; ++j)
        bs[j] += __shfl_xor(bs[j], mset, 32);
#pragma unroll
    for (int j = 0; j < 8; ++j) rsum[j] += bs[j];

    // write P (16q x 32k) bf16 tile to LDS (C layout -> row-major)
    {
      int col = lane & 15;
      int g = (lane >> 4) * 8;
#pragma unroll
      for (int j = 0; j < 8; ++j) {
        int m = j + g;
        L[m * 32 + col]      = f2bf(p0a[j]);
        L[m * 32 + 16 + col] = f2bf(p1a[j]);
      }
    }
    asm volatile("s_wait_dscnt 0x0" ::: "memory");

    // reload P as A fragment
    FragBF pf;
    {
      int r = lane & 15, c0 = (lane < 16) ? 0 : 8;
      const u16t* p = &L[r * 32 + c0];
      pf.q[0] = *(const v4u*)p;
      pf.q[1] = *(const v4u*)(p + 16);
    }
    // V fragments: B operand of P*V -> lane n is d-column, contiguous keys (vT layout)
    FragBF vf0, vf1;
    {
      int n = lane & 15, kb = (lane < 16) ? 0 : 8;
      const u16t* p0 = vbase + (size_t)n * SEQ + k0 + kb;
      vf0.q[0] = *(const v4u*)p0;
      vf0.q[1] = *(const v4u*)(p0 + 16);
      const u16t* p1 = vbase + (size_t)(16 + n) * SEQ + k0 + kb;
      vf1.q[0] = *(const v4u*)p1;
      vf1.q[1] = *(const v4u*)(p1 + 16);
    }
    o0 = wmma_bf16(pf, vf0, o0);
    o1 = wmma_bf16(pf, vf1, o1);
  }

  // normalize + store
  {
    int n = lane & 15;
    int g = (lane >> 4) * 8;
#pragma unroll
    for (int j = 0; j < 8; ++j) {
      int m = q0 + j + g;
      float inv = 1.0f / rsum[j];
      size_t base = ((size_t)(b * SEQ + m)) * MODEL + h * KSZ;
      o[base + n]      = f2bf(o0[j] * inv);
      o[base + 16 + n] = f2bf(o1[j] * inv);
    }
  }
}

// ---------- gating: top-1 over 2 experts ----------
__global__ __launch_bounds__(256) void vw_gate(const u16t* __restrict__ h,
                                               const float* __restrict__ wg,
                                               const float* __restrict__ bg,
                                               int* __restrict__ idx) {
  int t = blockIdx.x * 256 + threadIdx.x;
  if (t >= TOK_N) return;
  const u16t* hp = h + (size_t)t * MODEL;
  float g0 = bg[0], g1 = bg[1];
  for (int c = 0; c < MODEL; ++c) {
    float hv = bf2f(hp[c]);
    g0 += hv * wg[2 * c];
    g1 += hv * wg[2 * c + 1];
  }
  idx[t] = (g1 > g0) ? 1 : 0;   // argmax, first index wins ties
}

// ---------- expert input masking ----------
__global__ __launch_bounds__(256) void vw_mask(const u16t* __restrict__ h,
                                               const int* __restrict__ idx,
                                               int e, u16t* __restrict__ x) {
  int i = blockIdx.x * 256 + threadIdx.x;   // < TOK_N*MODEL
  x[i] = (idx[i >> 8] == e) ? h[i] : (u16t)0;
}

// ---------- final dense 128 -> 1 ----------
__global__ __launch_bounds__(256) void vw_final(const float* __restrict__ agg,
                                                const float* __restrict__ wd,
                                                const float* __restrict__ bd,
                                                float* __restrict__ out) {
  int t = blockIdx.x * 256 + threadIdx.x;
  if (t >= TOK_N) return;
  const float* a = agg + (size_t)t * 128;
  float s = bd[0];
#pragma unroll 4
  for (int c = 0; c < 128; ++c) s += a[c] * wd[c];
  out[t] = s;
}

// ---------- host orchestration ----------
static inline int gemm_blocks(int M, int N) { return ((M / 32) * (N / 64) + 7) / 8; }

extern "C" void kernel_launch(void* const* d_in, const int* in_sizes, int n_in,
                              void* d_out, int out_size, void* d_ws, size_t ws_size,
                              hipStream_t stream) {
  (void)in_sizes; (void)n_in; (void)out_size; (void)ws_size;

  // inputs (setup_inputs order)
  const int*   tok = (const int*)  d_in[0];
  const float* We  = (const float*)d_in[1];
  const float* wq  = (const float*)d_in[2];  const float* bq = (const float*)d_in[3];
  const float* wk  = (const float*)d_in[4];  const float* bk = (const float*)d_in[5];
  const float* wv  = (const float*)d_in[6];  const float* bv = (const float*)d_in[7];
  const float* wo  = (const float*)d_in[8];  const float* bo = (const float*)d_in[9];
  const float* w1  = (const float*)d_in[10]; const float* b1 = (const float*)d_in[11];
  const float* w2  = (const float*)d_in[12]; const float* b2 = (const float*)d_in[13];
  const float* wg  = (const float*)d_in[14]; const float* bg = (const float*)d_in[15];
  const float* ewq = (const float*)d_in[16]; const float* ebq = (const float*)d_in[17];
  const float* ewk = (const float*)d_in[18]; const float* ebk = (const float*)d_in[19];
  const float* ewv = (const float*)d_in[20]; const float* ebv = (const float*)d_in[21];
  const float* ewo = (const float*)d_in[22]; const float* ebo = (const float*)d_in[23];
  const float* ew1 = (const float*)d_in[24]; const float* eb1 = (const float*)d_in[25];
  const float* ew2 = (const float*)d_in[26]; const float* eb2 = (const float*)d_in[27];
  const float* wd  = (const float*)d_in[28]; const float* bd  = (const float*)d_in[29];

  // workspace layout
  char* ws = (char*)d_ws;
  size_t off = 0;
  auto take = [&](size_t bytes) -> void* {
    void* p = ws + off;
    off = (off + bytes + 255) & ~(size_t)255;
    return p;
  };

  u16t* wq_t  = (u16t*)take((size_t)128 * 256 * 2);
  u16t* wk_t  = (u16t*)take((size_t)128 * 256 * 2);
  u16t* wv_t  = (u16t*)take((size_t)128 * 256 * 2);
  u16t* wo_t  = (u16t*)take((size_t)256 * 256 * 2);
  u16t* w1_t  = (u16t*)take((size_t)256 * 512 * 2);
  u16t* w2_t  = (u16t*)take((size_t)512 * 256 * 2);
  u16t* ewq_t = (u16t*)take((size_t)2 * 256 * 256 * 2);
  u16t* ewk_t = (u16t*)take((size_t)2 * 256 * 256 * 2);
  u16t* ewv_t = (u16t*)take((size_t)2 * 256 * 256 * 2);
  u16t* ewo_t = (u16t*)take((size_t)2 * 256 * 256 * 2);
  u16t* ew1_t = (u16t*)take((size_t)2 * 256 * 256 * 2);
  u16t* ew2_t = (u16t*)take((size_t)2 * 256 * 128 * 2);

  u16t* emb   = (u16t*)take((size_t)TOK_N * 128 * 2);
  u16t* qb    = (u16t*)take((size_t)TOK_N * 256 * 2);
  u16t* kb    = (u16t*)take((size_t)TOK_N * 256 * 2);
  u16t* vTb   = (u16t*)take((size_t)TOK_N * 256 * 2);
  u16t* attno = (u16t*)take((size_t)TOK_N * 256 * 2);
  u16t* hob   = (u16t*)take((size_t)TOK_N * 256 * 2);
  u16t* h1b   = (u16t*)take((size_t)TOK_N * 512 * 2);
  u16t* hb    = (u16t*)take((size_t)TOK_N * 256 * 2);
  u16t* xb    = (u16t*)take((size_t)TOK_N * 256 * 2);
  float* agg  = (float*)take((size_t)TOK_N * 128 * 4);
  int*   idxb = (int*)  take((size_t)TOK_N * 4);

  auto tgrid = [](int K, int N) { return (K * N + 255) / 256; };

  // weight conversion (f32 [K,N] -> bf16 [N,K])
  vw_transw<<<tgrid(128, 256), 256, 0, stream>>>(wq, wq_t, 128, 256);
  vw_transw<<<tgrid(128, 256), 256, 0, stream>>>(wk, wk_t, 128, 256);
  vw_transw<<<tgrid(128, 256), 256, 0, stream>>>(wv, wv_t, 128, 256);
  vw_transw<<<tgrid(256, 256), 256, 0, stream>>>(wo, wo_t, 256, 256);
  vw_transw<<<tgrid(256, 512), 256, 0, stream>>>(w1, w1_t, 256, 512);
  vw_transw<<<tgrid(512, 256), 256, 0, stream>>>(w2, w2_t, 512, 256);
  for (int e = 0; e < 2; ++e) {
    vw_transw<<<tgrid(256, 256), 256, 0, stream>>>(ewq + e * 65536, ewq_t + e * 65536, 256, 256);
    vw_transw<<<tgrid(256, 256), 256, 0, stream>>>(ewk + e * 65536, ewk_t + e * 65536, 256, 256);
    vw_transw<<<tgrid(256, 256), 256, 0, stream>>>(ewv + e * 65536, ewv_t + e * 65536, 256, 256);
    vw_transw<<<tgrid(256, 256), 256, 0, stream>>>(ewo + e * 65536, ewo_t + e * 65536, 256, 256);
    vw_transw<<<tgrid(256, 256), 256, 0, stream>>>(ew1 + e * 65536, ew1_t + e * 65536, 256, 256);
    vw_transw<<<tgrid(256, 128), 256, 0, stream>>>(ew2 + e * 32768, ew2_t + e * 32768, 256, 128);
  }

  // embedding
  vw_embed<<<(TOK_N * 128) / 256, 256, 0, stream>>>(tok, We, emb);

  // main attention
  vw_gemm<<<gemm_blocks(TOK_N, 256), 256, 0, stream>>>(emb, wq_t, bq, qb,  TOK_N, 256, 128, 0);
  vw_gemm<<<gemm_blocks(TOK_N, 256), 256, 0, stream>>>(emb, wk_t, bk, kb,  TOK_N, 256, 128, 0);
  vw_gemm<<<gemm_blocks(TOK_N, 256), 256, 0, stream>>>(emb, wv_t, bv, vTb, TOK_N, 256, 128, 2);
  vw_attn<<<1024, 128, 0, stream>>>(qb, kb, vTb, attno);
  vw_gemm<<<gemm_blocks(TOK_N, 256), 256, 0, stream>>>(attno, wo_t, bo, hob, TOK_N, 256, 256, 0);
  vw_gemm<<<gemm_blocks(TOK_N, 512), 256, 0, stream>>>(hob, w1_t, b1, h1b, TOK_N, 512, 256, 0);
  vw_gemm<<<gemm_blocks(TOK_N, 256), 256, 0, stream>>>(h1b, w2_t, b2, hb,  TOK_N, 256, 512, 0);

  // gating
  vw_gate<<<TOK_N / 256, 256, 0, stream>>>(hb, wg, bg, idxb);

  // experts (accumulated into agg)
  hipMemsetAsync(agg, 0, (size_t)TOK_N * 128 * 4, stream);
  for (int e = 0; e < 2; ++e) {
    vw_mask<<<(TOK_N * 256) / 256, 256, 0, stream>>>(hb, idxb, e, xb);
    vw_gemm<<<gemm_blocks(TOK_N, 256), 256, 0, stream>>>(xb, ewq_t + e * 65536, ebq + e * 256, qb,  TOK_N, 256, 256, 0);
    vw_gemm<<<gemm_blocks(TOK_N, 256), 256, 0, stream>>>(xb, ewk_t + e * 65536, ebk + e * 256, kb,  TOK_N, 256, 256, 0);
    vw_gemm<<<gemm_blocks(TOK_N, 256), 256, 0, stream>>>(xb, ewv_t + e * 65536, ebv + e * 256, vTb, TOK_N, 256, 256, 2);
    vw_attn<<<1024, 128, 0, stream>>>(qb, kb, vTb, attno);
    vw_gemm<<<gemm_blocks(TOK_N, 256), 256, 0, stream>>>(attno, ewo_t + e * 65536, ebo + e * 256, hob, TOK_N, 256, 256, 0);
    vw_gemm<<<gemm_blocks(TOK_N, 256), 256, 0, stream>>>(hob, ew1_t + e * 65536, eb1 + e * 256, h1b, TOK_N, 256, 256, 0);
    vw_gemm<<<gemm_blocks(TOK_N, 128), 256, 0, stream>>>(h1b, ew2_t + e * 32768, eb2 + e * 128, agg, TOK_N, 128, 256, 1);
  }

  // final dense 128 -> 1
  vw_final<<<TOK_N / 256, 256, 0, stream>>>(agg, wd, bd, (float*)d_out);
}
